// CausalSelfAttention_17257178595936
// MI455X (gfx1250) — matrix operand order repaired and checked by reading, not compiled, not run on hardware
//
#include <hip/hip_runtime.h>

// ---------------------------------------------------------------------------
// CausalSelfAttention for MI455X (gfx1250, wave32, WMMA bf16, async->LDS)
// B=4, T=2048, C=1024, NH=16, HS=64
// ---------------------------------------------------------------------------

#define B_  4
#define T_  2048
#define C_  1024
#define NH_ 16
#define HS_ 64
#define M_  (B_ * T_)   // 8192 rows

typedef __attribute__((ext_vector_type(16))) __bf16 v16bf;
typedef __attribute__((ext_vector_type(8)))  float  v8f;

union Frag16 { uint4 u[2]; v16bf v; };
union U4BF   { uint4 u; unsigned short h[8]; };

__device__ __forceinline__ unsigned short f2bs(float f) {
  __bf16 h = (__bf16)f;
  return __builtin_bit_cast(unsigned short, h);
}
__device__ __forceinline__ float bs2f(unsigned short s) {
  return (float)__builtin_bit_cast(__bf16, s);
}
__device__ __forceinline__ v8f zero8() {
  v8f z;
#pragma unroll
  for (int i = 0; i < 8; ++i) z[i] = 0.f;
  return z;
}

// Flat shared-aperture addresses carry the LDS byte offset in the low 32 bits.
__device__ __forceinline__ unsigned ldsOff(const void* p) {
  return (unsigned)(uintptr_t)p;
}
// Async global -> LDS copy of 16 bytes per lane (ASYNCcnt-tracked, no VGPR data).
__device__ __forceinline__ void asyncCopy16(unsigned ldsAddr, const void* gsrc) {
  unsigned long long ga = (unsigned long long)(uintptr_t)gsrc;
  asm volatile("global_load_async_to_lds_b128 %0, %1, off"
               :: "v"(ldsAddr), "v"(ga) : "memory");
}
__device__ __forceinline__ void waitAsync0() {
  asm volatile("s_wait_asynccnt 0x0" ::: "memory");
}

// 16-bit A-matrix 16x32 fragment (ISA 7.12.2): lanes 0-15 hold K=0..7,16..23,
// lanes 16-31 hold K=8..15,24..31 for row M = lane%16. rowBase points at K=0.
__device__ __forceinline__ v16bf ldA32(const unsigned short* rowBase, int lane) {
  const int koff = (lane & 16) ? 8 : 0;
  Frag16 f;
  f.u[0] = *reinterpret_cast<const uint4*>(rowBase + koff);
  f.u[1] = *reinterpret_cast<const uint4*>(rowBase + 16 + koff);
  return f.v;
}
// 16-bit B-matrix 32x16 fragment: column N = lane%16; lanes 0-15 hold K=0..7,16..23,
// lanes 16-31 hold K=8..15,24..31 (same as A). colBase points at K=0 of the column.
__device__ __forceinline__ v16bf ldB32(const unsigned short* colBase, int lane) {
  const int koff = (lane & 16) ? 8 : 0;
  Frag16 f;
  f.u[0] = *reinterpret_cast<const uint4*>(colBase + koff);
  f.u[1] = *reinterpret_cast<const uint4*>(colBase + 16 + koff);
  return f.v;
}
__device__ __forceinline__ v8f wmma_bf16(v16bf a, v16bf b, v8f c) {
  return __builtin_amdgcn_wmma_f32_16x16x32_bf16(false, a, false, b,
                                                 (short)0, c, false, false);
}

// ---------------------------------------------------------------------------
// fp32 -> bf16 conversion
// ---------------------------------------------------------------------------
__global__ void cvt_bf16_kernel(const float* __restrict__ src,
                                unsigned short* __restrict__ dst, int n) {
  int tid = blockIdx.x * blockDim.x + threadIdx.x;
  if (tid < n) dst[tid] = f2bs(src[tid]);
}

// ---------------------------------------------------------------------------
// RoPE cos/sin tables: [T_][HS_/2]
// ---------------------------------------------------------------------------
__global__ void rope_table_kernel(float* __restrict__ cosT,
                                  float* __restrict__ sinT) {
  int tid = blockIdx.x * blockDim.x + threadIdx.x;
  if (tid >= T_ * (HS_ / 2)) return;
  int t = tid >> 5, j = tid & 31;
  float inv = __powf(10000.f, -(2.f * (float)j) / (float)HS_);
  float ang = (float)t * inv;
  cosT[tid] = __cosf(ang);
  sinT[tid] = __sinf(ang);
}

// ---------------------------------------------------------------------------
// In-place RoPE on q and k, layout [B*NH, T, HS] bf16
// ---------------------------------------------------------------------------
__global__ void rope_apply_kernel(unsigned short* __restrict__ q,
                                  unsigned short* __restrict__ k,
                                  const float* __restrict__ cosT,
                                  const float* __restrict__ sinT) {
  int tid = blockIdx.x * blockDim.x + threadIdx.x;   // B_*NH_*T_*32
  if (tid >= B_ * NH_ * T_ * (HS_ / 2)) return;
  int j  = tid & 31;
  int t  = (tid >> 5) & (T_ - 1);
  int bh = tid >> 16;                                // T_*32 == 65536
  size_t base = ((size_t)bh * T_ + t) * HS_ + 2 * j;
  float c = cosT[t * 32 + j], s = sinT[t * 32 + j];
  float q0 = bs2f(q[base]), q1 = bs2f(q[base + 1]);
  q[base]     = f2bs(q0 * c - q1 * s);
  q[base + 1] = f2bs(q0 * s + q1 * c);
  float k0 = bs2f(k[base]), k1 = bs2f(k[base + 1]);
  k[base]     = f2bs(k0 * c - k1 * s);
  k[base + 1] = f2bs(k0 * s + k1 * c);
}

// ---------------------------------------------------------------------------
// bf16 WMMA GEMM: out = A[M_,C_] @ W[C_,C_] + bias
// Block tile 128(M) x 128(N), K-step 32, double-buffered LDS.
// 8 waves in a 4(M) x 2(N) grid: wave computes 32 rows x 64 cols = 8 accums.
// All 6 fragments are loaded into distinct registers before the WMMA chain so
// the 8 WMMAs issue back-to-back instead of being fenced by s_wait_dscnt 0.
// A tile staged with GLOBAL_LOAD_ASYNC_TO_LDS_B128; W tile transposed via VGPR.
// splitHead=1: write bf16 to [B,NH,T,HS]; splitHead=0: write fp32 [M_,C_].
// ---------------------------------------------------------------------------
#define ASTR 40   // LDS row stride (bf16 elems), 80B keeps 16B alignment
__global__ __launch_bounds__(256) void gemm_bf16_kernel(
    const unsigned short* __restrict__ A,
    const unsigned short* __restrict__ W,
    const float* __restrict__ bias,
    void* __restrict__ outp,
    int splitHead) {
  __shared__ __align__(16) unsigned short Asm[2][128 * ASTR];   // [m][k]
  __shared__ __align__(16) unsigned short Bsm[2][128 * ASTR];   // [n][k] (transposed)
  const int tid   = threadIdx.x;
  const int lane  = tid & 31, wave = tid >> 5;
  const int waveM = wave >> 1, waveN = wave & 1;
  const int nBase = blockIdx.x * 128;
  const int mBase = blockIdx.y * 128;

  v8f acc[2][4];
#pragma unroll
  for (int s = 0; s < 2; ++s)
#pragma unroll
    for (int n = 0; n < 4; ++n) acc[s][n] = zero8();

  auto loadTile = [&](int kk, int buf) {
    // A tile 128x32: async global->LDS, 16B per lane, 2 issues per thread.
#pragma unroll
    for (int i = 0; i < 2; ++i) {
      int idx = tid + i * 256;              // 0..511
      int row = idx >> 2, c8 = (idx & 3) * 8;
      asyncCopy16(ldsOff(&Asm[buf][row * ASTR + c8]),
                  &A[(size_t)(mBase + row) * C_ + kk + c8]);
    }
    // W tile 32x128, stored transposed into Bsm[buf][n][k].
#pragma unroll
    for (int i = 0; i < 2; ++i) {
      int idx = tid + i * 256;              // 0..511
      int r = idx >> 4, c8 = (idx & 15) * 8;
      U4BF u;
      u.u = *reinterpret_cast<const uint4*>(&W[(size_t)(kk + r) * C_ + nBase + c8]);
#pragma unroll
      for (int j = 0; j < 8; ++j) Bsm[buf][(c8 + j) * ASTR + r] = u.h[j];
    }
  };

  loadTile(0, 0);
  waitAsync0();
  __syncthreads();

  int p = 0;
  for (int kk = 0; kk < C_; kk += 32) {
    if (kk + 32 < C_) loadTile(kk + 32, p ^ 1);   // overlap with WMMA below

    // Load ALL fragments first (distinct registers), then the WMMA chain.
    v16bf af0 = ldA32(&Asm[p][(waveM * 32 + (lane & 15)) * ASTR], lane);
    v16bf af1 = ldA32(&Asm[p][(waveM * 32 + 16 + (lane & 15)) * ASTR], lane);
    v16bf bfr[4];
#pragma unroll
    for (int n = 0; n < 4; ++n)
      bfr[n] = ldB32(&Bsm[p][(waveN * 64 + n * 16 + (lane & 15)) * ASTR], lane);
#pragma unroll
    for (int n = 0; n < 4; ++n) {
      acc[0][n] = wmma_bf16(af0, bfr[n], acc[0][n]);
      acc[1][n] = wmma_bf16(af1, bfr[n], acc[1][n]);
    }
    waitAsync0();          // next tile's async A writes landed
    __syncthreads();       // everyone done reading buf p / writing buf p^1
    p ^= 1;
  }

  // Epilogue. D layout: vgpr i, lanes 0-15 -> M=i, lanes 16-31 -> M=8+i; N=lane%16.
  const int half8 = (lane & 16) ? 8 : 0;
#pragma unroll
  for (int sm = 0; sm < 2; ++sm) {
    const int mrowBase = mBase + waveM * 32 + sm * 16 + half8;
#pragma unroll
    for (int n = 0; n < 4; ++n) {
      int c = nBase + waveN * 64 + n * 16 + (lane & 15);
      float bv = bias[c];
#pragma unroll
      for (int i = 0; i < 8; ++i) {
        int mg = mrowBase + i;
        float val = acc[sm][n][i] + bv;
        if (splitHead) {
          int b = mg >> 11, t = mg & (T_ - 1);
          int h = c >> 6, d = c & (HS_ - 1);
          reinterpret_cast<unsigned short*>(outp)
              [(((size_t)(b * NH_ + h)) * T_ + t) * HS_ + d] = f2bs(val);
        } else {
          reinterpret_cast<float*>(outp)[(size_t)mg * C_ + c] = val;
        }
      }
    }
  }
}

// ---------------------------------------------------------------------------
// Flash attention: grid (T_/128 q-tiles, B_*NH_). 8 waves; wave owns 16 q rows.
// K tiles staged async into LDS; V staged transposed; online softmax; P goes
// through per-wave LDS to convert D-layout -> A-layout for the P@V WMMAs.
// ---------------------------------------------------------------------------
#define KSTR 72   // 144B row stride
__global__ __launch_bounds__(256) void attn_kernel(
    const unsigned short* __restrict__ Q,
    const unsigned short* __restrict__ K,
    const unsigned short* __restrict__ V,
    unsigned short* __restrict__ Y) {
  __shared__ __align__(16) unsigned short Ksm[64 * KSTR];        // [key][d]
  __shared__ __align__(16) unsigned short Vsm[64 * KSTR];        // [d][key]
  __shared__ __align__(16) unsigned short Psm[8 * 16 * KSTR];    // per-wave [row][key]
  const int tid = threadIdx.x, lane = tid & 31, wave = tid >> 5;
  const int qBase = blockIdx.x * 128;
  const int bh = blockIdx.y;
  const unsigned short* Qb = Q + (size_t)bh * T_ * HS_;
  const unsigned short* Kb = K + (size_t)bh * T_ * HS_;
  const unsigned short* Vb = V + (size_t)bh * T_ * HS_;

  // Q fragments stay in registers for the whole kernel (16 rows x 64 d).
  const int qrow = qBase + wave * 16 + (lane & 15);
  const unsigned short* qp = Qb + (size_t)qrow * HS_;
  v16bf aq0 = ldA32(qp, lane);
  v16bf aq1 = ldA32(qp + 32, lane);

  v8f o[4];
#pragma unroll
  for (int n = 0; n < 4; ++n) o[n] = zero8();
  float mrow[8], lrow[8];
#pragma unroll
  for (int i = 0; i < 8; ++i) { mrow[i] = -1e30f; lrow[i] = 0.f; }

  unsigned short* Pw = &Psm[wave * 16 * KSTR];
  const float scale = 0.125f;                 // 1/sqrt(64)
  const int nkt = (qBase >> 6) + 2;           // causal: keys up to qBase+127

  for (int kt = 0; kt < nkt; ++kt) {
    const int kBase = kt * 64;
    // K tile: async global->LDS. V tile: transpose via VGPR into [d][key].
#pragma unroll
    for (int i = 0; i < 2; ++i) {
      int idx = tid + i * 256;                // 0..511
      int row = idx >> 3, c8 = (idx & 7) * 8;
      asyncCopy16(ldsOff(&Ksm[row * KSTR + c8]),
                  &Kb[(size_t)(kBase + row) * HS_ + c8]);
      U4BF uv;
      uv.u = *reinterpret_cast<const uint4*>(&Vb[(size_t)(kBase + row) * HS_ + c8]);
#pragma unroll
      for (int j = 0; j < 8; ++j) Vsm[(c8 + j) * KSTR + row] = uv.h[j];
    }
    waitAsync0();
    __syncthreads();

    // S = Q @ K^T : 16 q rows x 64 keys per wave (contraction over d = 2 WMMAs).
    v8f s[4];
#pragma unroll
    for (int n = 0; n < 4; ++n) {
      const unsigned short* kb = &Ksm[(n * 16 + (lane & 15)) * KSTR];
      v16bf b0 = ldB32(kb, lane);
      v16bf b1 = ldB32(kb + 32, lane);
      s[n] = wmma_bf16(aq0, b0, zero8());
      s[n] = wmma_bf16(aq1, b1, s[n]);
    }

    // Scale + causal mask + online softmax (rows live in 16-lane halves).
    const int keyc = kBase + (lane & 15);
    const int qr0 = qBase + wave * 16 + ((lane & 16) ? 8 : 0);
    float tmax[8];
#pragma unroll
    for (int i = 0; i < 8; ++i) {
      int qg = qr0 + i;
      float vmax = -1e30f;
#pragma unroll
      for (int n = 0; n < 4; ++n) {
        float sv = s[n][i] * scale;
        int key = keyc + n * 16;
        sv = (key > qg) ? -1e30f : sv;
        s[n][i] = sv;
        vmax = fmaxf(vmax, sv);
      }
      tmax[i] = vmax;
    }
#pragma unroll
    for (int i = 0; i < 8; ++i) {
      float v = tmax[i];
      v = fmaxf(v, __shfl_xor(v, 1, 32));
      v = fmaxf(v, __shfl_xor(v, 2, 32));
      v = fmaxf(v, __shfl_xor(v, 4, 32));
      v = fmaxf(v, __shfl_xor(v, 8, 32));
      float mnew = fmaxf(mrow[i], v);
      float corr = __expf(mrow[i] - mnew);
      float rsum = 0.f;
#pragma unroll
      for (int n = 0; n < 4; ++n) {
        float pv = __expf(s[n][i] - mnew);
        s[n][i] = pv;
        rsum += pv;
      }
      rsum += __shfl_xor(rsum, 1, 32);
      rsum += __shfl_xor(rsum, 2, 32);
      rsum += __shfl_xor(rsum, 4, 32);
      rsum += __shfl_xor(rsum, 8, 32);
      lrow[i] = lrow[i] * corr + rsum;
      mrow[i] = mnew;
#pragma unroll
      for (int n = 0; n < 4; ++n) o[n][i] *= corr;
    }

    // P (D layout) -> per-wave LDS [row][key] as bf16.
#pragma unroll
    for (int n = 0; n < 4; ++n) {
      int key = n * 16 + (lane & 15);
      int rb = (lane & 16) ? 8 : 0;
#pragma unroll
      for (int i = 0; i < 8; ++i)
        Pw[(rb + i) * KSTR + key] = f2bs(s[n][i]);
    }
    // Same-wave LDS store->load ordering; force data visibility.
    asm volatile("s_wait_dscnt 0x0" ::: "memory");

    // O += P @ V (contraction over 64 keys = 2 WMMAs per 16-wide d tile).
    const unsigned short* prow = &Pw[(lane & 15) * KSTR];
    v16bf ap0 = ldA32(prow, lane);
    v16bf ap1 = ldA32(prow + 32, lane);
#pragma unroll
    for (int n = 0; n < 4; ++n) {
      const unsigned short* vb = &Vsm[(n * 16 + (lane & 15)) * KSTR];
      v16bf bv0 = ldB32(vb, lane);
      v16bf bv1 = ldB32(vb + 32, lane);
      o[n] = wmma_bf16(ap0, bv0, o[n]);
      o[n] = wmma_bf16(ap1, bv1, o[n]);
    }
    __syncthreads();   // protect Ksm/Vsm before next tile's loads
  }

  // Normalize and store to y [B,T,C] bf16.
  const int b = bh >> 4, h = bh & 15;
  const int rb = qBase + wave * 16 + ((lane & 16) ? 8 : 0);
#pragma unroll
  for (int n = 0; n < 4; ++n) {
    int c = h * HS_ + n * 16 + (lane & 15);
#pragma unroll
    for (int i = 0; i < 8; ++i) {
      float val = o[n][i] / lrow[i];
      Y[((size_t)(b * T_ + rb + i)) * C_ + c] = f2bs(val);
    }
  }
}

// ---------------------------------------------------------------------------
// Host launcher
// ---------------------------------------------------------------------------
extern "C" void kernel_launch(void* const* d_in, const int* in_sizes, int n_in,
                              void* d_out, int out_size, void* d_ws, size_t ws_size,
                              hipStream_t stream) {
  (void)in_sizes; (void)n_in; (void)out_size; (void)ws_size;
  const float* x  = (const float*)d_in[0];
  const float* Wq = (const float*)d_in[1];
  const float* bq = (const float*)d_in[2];
  const float* Wk = (const float*)d_in[3];
  const float* bk = (const float*)d_in[4];
  const float* Wv = (const float*)d_in[5];
  const float* bv = (const float*)d_in[6];
  const float* Wp = (const float*)d_in[7];
  const float* bp = (const float*)d_in[8];

  char* ws = (char*)d_ws;
  size_t off = 0;
  const size_t xbBytes = (size_t)M_ * C_ * 2;     // 16 MB
  const size_t wBytes  = (size_t)C_ * C_ * 2;     // 2 MB
  unsigned short* xb   = (unsigned short*)(ws + off); off += xbBytes;
  unsigned short* Wqb  = (unsigned short*)(ws + off); off += wBytes;
  unsigned short* Wkb  = (unsigned short*)(ws + off); off += wBytes;
  unsigned short* Wvb  = (unsigned short*)(ws + off); off += wBytes;
  unsigned short* Wpb  = (unsigned short*)(ws + off); off += wBytes;
  unsigned short* qbuf = (unsigned short*)(ws + off); off += xbBytes;
  unsigned short* kbuf = (unsigned short*)(ws + off); off += xbBytes;
  unsigned short* vbuf = (unsigned short*)(ws + off); off += xbBytes;
  unsigned short* ybf  = (unsigned short*)(ws + off); off += xbBytes;
  float* cosT = (float*)(ws + off); off += (size_t)T_ * (HS_ / 2) * 4;
  float* sinT = (float*)(ws + off); off += (size_t)T_ * (HS_ / 2) * 4;

  // 1) fp32 -> bf16
  {
    int n = M_ * C_;
    cvt_bf16_kernel<<<n / 256, 256, 0, stream>>>(x, xb, n);
    int nw = C_ * C_;
    cvt_bf16_kernel<<<nw / 256, 256, 0, stream>>>(Wq, Wqb, nw);
    cvt_bf16_kernel<<<nw / 256, 256, 0, stream>>>(Wk, Wkb, nw);
    cvt_bf16_kernel<<<nw / 256, 256, 0, stream>>>(Wv, Wvb, nw);
    cvt_bf16_kernel<<<nw / 256, 256, 0, stream>>>(Wp, Wpb, nw);
  }
  // 2) RoPE tables
  rope_table_kernel<<<(T_ * 32) / 256, 256, 0, stream>>>(cosT, sinT);

  // 3) QKV projections (WMMA GEMM), split-head bf16 outputs
  dim3 gGrid(C_ / 128, M_ / 128);
  gemm_bf16_kernel<<<gGrid, 256, 0, stream>>>(xb, Wqb, bq, qbuf, 1);
  gemm_bf16_kernel<<<gGrid, 256, 0, stream>>>(xb, Wkb, bk, kbuf, 1);
  gemm_bf16_kernel<<<gGrid, 256, 0, stream>>>(xb, Wvb, bv, vbuf, 1);

  // 4) RoPE on q, k
  {
    int n = B_ * NH_ * T_ * (HS_ / 2);
    rope_apply_kernel<<<n / 256, 256, 0, stream>>>(qbuf, kbuf, cosT, sinT);
  }

  // 5) Flash attention -> y bf16 [B,T,C]
  dim3 aGrid(T_ / 128, B_ * NH_);
  attn_kernel<<<aGrid, 256, 0, stream>>>(qbuf, kbuf, vbuf, ybf);

  // 6) Output projection -> fp32 d_out
  gemm_bf16_kernel<<<gGrid, 256, 0, stream>>>(ybf, Wpb, bp, d_out, 0);
}
